// AttentionPoolingAdvance_53566832116145
// MI455X (gfx1250) — matrix operand order repaired and checked by reading, not compiled, not run on hardware
//
#include <hip/hip_runtime.h>
#include <hip/hip_bf16.h>

typedef __bf16 bf16_t;
typedef bf16_t v16bf __attribute__((ext_vector_type(16)));
typedef bf16_t v8bf  __attribute__((ext_vector_type(8)));
typedef float  v8f   __attribute__((ext_vector_type(8)));
typedef int    v4i   __attribute__((ext_vector_type(4)));

// Address-space qualified element types for the async global->LDS builtin:
// param1 is AS1 (global, printed as __device__), param2 is AS3 (LDS).
typedef __attribute__((address_space(1))) v4i v4i_g;
typedef __attribute__((address_space(3))) v4i v4i_l;

#define D_DIM   768
#define S_SEQ   2048
#define B_BATCH 8
#define INV_SCALE 0.03608439182435161f   // 1/sqrt(768)
#define INV_S     (1.0f/2048.0f)

#ifndef __has_builtin
#define __has_builtin(x) 0
#endif

// ---- CDNA5 async global->LDS path (ASYNCcnt), with safe fallback ----------
#if __has_builtin(__builtin_amdgcn_global_load_async_to_lds_b128)
#define HAVE_ASYNC_LDS 1
#else
#define HAVE_ASYNC_LDS 0
#endif

#if HAVE_ASYNC_LDS
#if __has_builtin(__builtin_amdgcn_s_wait_asynccnt)
#define WAIT_ASYNC_12() __builtin_amdgcn_s_wait_asynccnt(12)
#define WAIT_ASYNC_0()  __builtin_amdgcn_s_wait_asynccnt(0)
#else
#define WAIT_ASYNC_12() asm volatile("s_wait_asynccnt 0xc" ::: "memory")
#define WAIT_ASYNC_0()  asm volatile("s_wait_asynccnt 0x0" ::: "memory")
#endif
#else
#define WAIT_ASYNC_12() ((void)0)
#define WAIT_ASYNC_0()  ((void)0)
#endif

// Stage one 16x768 bf16 tile (24KB = 1536 x 16B chunks; 12 chunks/thread @128 thr)
__device__ __forceinline__ void stage_tile(const bf16_t* __restrict__ src,
                                           bf16_t* __restrict__ dst, int tid) {
#if HAVE_ASYNC_LDS
#pragma unroll
    for (int i = 0; i < 12; ++i) {
        const int idx = tid + i * 128;
        __builtin_amdgcn_global_load_async_to_lds_b128(
            (v4i_g*)(src + idx * 8),
            (v4i_l*)(dst + idx * 8), 0, 0);
    }
#else
    const uint4* s4 = reinterpret_cast<const uint4*>(src);
    uint4* d4 = reinterpret_cast<uint4*>(dst);
    for (int i = tid; i < 16 * D_DIM / 8; i += 128) d4[i] = s4[i];
#endif
}

union AFrag { v16bf v; v8bf h[2]; };

// A fragment (16x32 bf16, wave32): lane L<16 -> row M=L, K = d0+0..15;
// lane L>=16 -> row M=L-16, K = d0+16..31. Two 16B contiguous LDS loads.
__device__ __forceinline__ v16bf load_a_frag(const bf16_t* base, int row0, int d0, int lane) {
    const int arow = lane & 15;
    const int asel = (lane >> 4) * 16;
    const bf16_t* ptr = base + (size_t)(row0 + arow) * D_DIM + d0 + asel;
    AFrag f;
    f.h[0] = *reinterpret_cast<const v8bf*>(ptr);
    f.h[1] = *reinterpret_cast<const v8bf*>(ptr + 8);
    return f.v;
}

// B fragment (32x16 bf16, wave32): lane L<16 -> col N=L, K = d0+0..15;
// lane L>=16 -> col N=L-16, K = d0+16..31. One 32B contiguous LDS load.
__device__ __forceinline__ v16bf load_b_frag(const bf16_t* base, int d0, int lane) {
    const int bcol = lane & 15;
    const int bsel = (lane >> 4) * 16;
    const bf16_t* ptr = base + (size_t)bcol * D_DIM + d0 + bsel;
    return *reinterpret_cast<const v16bf*>(ptr);
}

__global__ void k_cvt(const float* __restrict__ in, bf16_t* __restrict__ out, int n) {
    int i = blockIdx.x * blockDim.x + threadIdx.x;
    int stride = gridDim.x * blockDim.x;
    for (; i < n; i += stride) out[i] = (bf16_t)in[i];
}

__global__ void k_zero(float* __restrict__ p, int n) {
    int i = blockIdx.x * blockDim.x + threadIdx.x;
    if (i < n) p[i] = 0.0f;
}

// QKV projection. 4 waves/WG, each owns a 16-row s-subtile of a 64-row X block.
// X block (96KB) resident in LDS; W tiles (24KB) double-buffered + async-staged.
__global__ __launch_bounds__(128)
void k_qkv(const bf16_t* __restrict__ xb,
           const bf16_t* __restrict__ wq, const bf16_t* __restrict__ wk,
           const bf16_t* __restrict__ wv,
           const float* __restrict__ bq, const float* __restrict__ bk,
           const float* __restrict__ bv,
           bf16_t* __restrict__ Qb, bf16_t* __restrict__ Kb, float* __restrict__ Vf) {
    extern __shared__ char smem[];
    bf16_t* Xs  = (bf16_t*)smem;                 // 64 x 768
    bf16_t* Ws0 = Xs + 64 * D_DIM;               // 2 x (16 x 768)

    const int tid  = threadIdx.x;
    const int lane = tid & 31;
    const int wave = tid >> 5;
    const int b  = blockIdx.x >> 5;              // 32 blocks per batch (S/64)
    const int s0 = (blockIdx.x & 31) * 64;

    const int NTILE = 3 * (D_DIM / 16);          // 144

    // W-tile base pointer for flattened tile index t
    auto wtile = [&](int t) -> const bf16_t* {
        const int which = t / (D_DIM / 16);
        const int o0 = (t % (D_DIM / 16)) * 16;
        const bf16_t* w = (which == 0) ? wq : ((which == 1) ? wk : wv);
        return w + (size_t)o0 * D_DIM;
    };

    // prologue: stage first W tile async, stage X block sync
    stage_tile(wtile(0), Ws0, tid);
    {
        const uint4* src = reinterpret_cast<const uint4*>(xb + ((size_t)b * S_SEQ + s0) * D_DIM);
        uint4* dst = reinterpret_cast<uint4*>(Xs);
        for (int i = tid; i < 64 * D_DIM / 8; i += 128) dst[i] = src[i];
    }

    for (int t = 0; t < NTILE; ++t) {
        const int which = t / (D_DIM / 16);
        const int o0 = (t % (D_DIM / 16)) * 16;
        const float* bsel = (which == 0) ? bq : ((which == 1) ? bk : bv);

        __syncthreads();                          // readers of buffer (t+1)&1 done
        if (t + 1 < NTILE) {
            stage_tile(wtile(t + 1), Ws0 + ((t + 1) & 1) * 16 * D_DIM, tid);
            if (t + 2 < NTILE) __builtin_prefetch(wtile(t + 2) + tid * 96, 0, 1);
            WAIT_ASYNC_12();                      // current tile (issued last iter) landed
        } else {
            WAIT_ASYNC_0();
        }
        __syncthreads();

        const bf16_t* Ws = Ws0 + (t & 1) * 16 * D_DIM;
        v8f c = {};
#pragma unroll
        for (int d0 = 0; d0 < D_DIM; d0 += 32) {
            v16bf a  = load_a_frag(Xs, wave * 16, d0, lane);
            v16bf bb = load_b_frag(Ws, d0, lane);
            c = __builtin_amdgcn_wmma_f32_16x16x32_bf16(false, a, false, bb,
                                                        (short)0, c, false, false);
        }

        const float bias = bsel[o0 + (lane & 15)];
        const int col = lane & 15;
        const int rhi = (lane >> 4) * 8;          // C layout: row = r + 8*(lane>=16)
        const size_t rowbase = (size_t)b * S_SEQ + s0 + wave * 16;
        if (which == 2) {
#pragma unroll
            for (int r = 0; r < 8; ++r)
                Vf[(rowbase + rhi + r) * D_DIM + o0 + col] = c[r] + bias;
        } else {
            bf16_t* dst = (which == 0) ? Qb : Kb;
#pragma unroll
            for (int r = 0; r < 8; ++r)
                dst[(rowbase + rhi + r) * D_DIM + o0 + col] = (bf16_t)(c[r] + bias);
        }
    }
}

// Attention pooling. Each WG owns 64 query rows (Q block resident in LDS).
// Flattened 2*128 tile loop: phase 0 = online row max/sumexp, phase 1 =
// recompute scores + atomically accumulate pooled key weights p[b][k].
// K tiles double-buffered + async-staged.
__global__ __launch_bounds__(128)
void k_attn(const bf16_t* __restrict__ Qb, const bf16_t* __restrict__ Kb,
            const int* __restrict__ mask, float* __restrict__ p) {
    extern __shared__ char smem[];
    bf16_t* Qs  = (bf16_t*)smem;                 // 64 x 768
    bf16_t* Ks0 = Qs + 64 * D_DIM;               // 2 x (16 x 768)

    const int tid  = threadIdx.x;
    const int lane = tid & 31;
    const int wave = tid >> 5;
    const int b  = blockIdx.x >> 5;
    const int q0 = (blockIdx.x & 31) * 64;

    const int NT = S_SEQ / 16;                   // 128

    // prologue: first K tile async, Q block sync
    stage_tile(Kb + (size_t)b * S_SEQ * D_DIM, Ks0, tid);
    {
        const uint4* src = reinterpret_cast<const uint4*>(Qb + ((size_t)b * S_SEQ + q0) * D_DIM);
        uint4* dst = reinterpret_cast<uint4*>(Qs);
        for (int i = tid; i < 64 * D_DIM / 8; i += 128) dst[i] = src[i];
    }

    float m[8], l[8], rinv[8];
#pragma unroll
    for (int r = 0; r < 8; ++r) { m[r] = -1e30f; l[r] = 0.0f; rinv[r] = 0.0f; }

    for (int j = 0; j < 2 * NT; ++j) {
        const int kt = j & (NT - 1);
        const int phase = j >> 7;                // NT == 128

        __syncthreads();                          // readers of buffer (j+1)&1 done
        if (j + 1 < 2 * NT) {
            const int nk = (j + 1) & (NT - 1);
            stage_tile(Kb + ((size_t)b * S_SEQ + nk * 16) * D_DIM,
                       Ks0 + ((j + 1) & 1) * 16 * D_DIM, tid);
            const int pk2 = (j + 2) & (NT - 1);
            __builtin_prefetch(Kb + ((size_t)b * S_SEQ + pk2 * 16) * D_DIM + tid * 96, 0, 1);
            WAIT_ASYNC_12();
        } else {
            WAIT_ASYNC_0();
        }
        __syncthreads();

        const bf16_t* Ks = Ks0 + (j & 1) * 16 * D_DIM;
        v8f c = {};
#pragma unroll
        for (int d0 = 0; d0 < D_DIM; d0 += 32) {
            v16bf a  = load_a_frag(Qs, wave * 16, d0, lane);
            v16bf bb = load_b_frag(Ks, d0, lane);
            c = __builtin_amdgcn_wmma_f32_16x16x32_bf16(false, a, false, bb,
                                                        (short)0, c, false, false);
        }

        const bool masked = (mask[b * S_SEQ + kt * 16 + (lane & 15)] == 0);
        if (phase == 0) {
#pragma unroll
            for (int r = 0; r < 8; ++r) {
                float s  = masked ? -1e30f : c[r] * INV_SCALE;
                float nm = fmaxf(m[r], s);
                l[r] = l[r] * __expf(m[r] - nm) + __expf(s - nm);
                m[r] = nm;
            }
            if (kt == NT - 1) {
                // combine (m,l) across the 16 lanes holding the same row
#pragma unroll
                for (int off = 1; off <= 8; off <<= 1) {
#pragma unroll
                    for (int r = 0; r < 8; ++r) {
                        float mo = __shfl_xor(m[r], off, 32);
                        float lo = __shfl_xor(l[r], off, 32);
                        float nm = fmaxf(m[r], mo);
                        l[r] = l[r] * __expf(m[r] - nm) + lo * __expf(mo - nm);
                        m[r] = nm;
                    }
                }
#pragma unroll
                for (int r = 0; r < 8; ++r) rinv[r] = 1.0f / l[r];
            }
        } else if (!masked) {
            float psum = 0.0f;
#pragma unroll
            for (int r = 0; r < 8; ++r)
                psum += __expf(c[r] * INV_SCALE - m[r]) * rinv[r];
            atomicAdd(&p[b * S_SEQ + kt * 16 + (lane & 15)], psum * INV_S);
        }
    }
}

// out[b,0,o] = sum_k p[b,k] * V[b,k,o]
__global__ __launch_bounds__(256)
void k_final(const float* __restrict__ p, const float* __restrict__ Vf,
             float* __restrict__ out) {
    const int b = blockIdx.x;
    const int o = threadIdx.x;
    float a0 = 0.0f, a1 = 0.0f, a2 = 0.0f;
    const float* pb = p + b * S_SEQ;
    const float* vb = Vf + (size_t)b * S_SEQ * D_DIM;
    for (int k = 0; k < S_SEQ; ++k) {
        const float pk = pb[k];
        const float* vr = vb + (size_t)k * D_DIM;
        a0 += pk * vr[o];
        a1 += pk * vr[o + 256];
        a2 += pk * vr[o + 512];
    }
    out[b * D_DIM + o]       = a0;
    out[b * D_DIM + o + 256] = a1;
    out[b * D_DIM + o + 512] = a2;
}

extern "C" void kernel_launch(void* const* d_in, const int* in_sizes, int n_in,
                              void* d_out, int out_size, void* d_ws, size_t ws_size,
                              hipStream_t stream) {
    const float* x    = (const float*)d_in[0];
    const int*   mask = (const int*)d_in[1];
    const float* Wq   = (const float*)d_in[2];
    const float* bq   = (const float*)d_in[3];
    const float* Wk   = (const float*)d_in[4];
    const float* bk   = (const float*)d_in[5];
    const float* Wv   = (const float*)d_in[6];
    const float* bv   = (const float*)d_in[7];
    float* out = (float*)d_out;

    char* ws = (char*)d_ws;
    size_t off = 0;
    const size_t xsz = (size_t)B_BATCH * S_SEQ * D_DIM;
    const size_t wsz = (size_t)D_DIM * D_DIM;

    bf16_t* xb  = (bf16_t*)(ws + off); off += xsz * 2;
    bf16_t* wqb = (bf16_t*)(ws + off); off += wsz * 2;
    bf16_t* wkb = (bf16_t*)(ws + off); off += wsz * 2;
    bf16_t* wvb = (bf16_t*)(ws + off); off += wsz * 2;
    bf16_t* Qb  = (bf16_t*)(ws + off); off += xsz * 2;
    bf16_t* Kb  = (bf16_t*)(ws + off); off += xsz * 2;
    float*  Vf  = (float*)(ws + off);  off += xsz * 4;
    float*  p   = (float*)(ws + off);  off += (size_t)B_BATCH * S_SEQ * 4;
    (void)ws_size; (void)in_sizes; (void)n_in; (void)out_size;

    const int ldsBytes = (64 + 2 * 16) * D_DIM * 2;  // 144 KB
    (void)hipFuncSetAttribute(reinterpret_cast<const void*>(k_qkv),
                              hipFuncAttributeMaxDynamicSharedMemorySize, ldsBytes);
    (void)hipFuncSetAttribute(reinterpret_cast<const void*>(k_attn),
                              hipFuncAttributeMaxDynamicSharedMemorySize, ldsBytes);

    k_cvt<<<2048, 256, 0, stream>>>(x,  xb,  (int)xsz);
    k_cvt<<<512,  256, 0, stream>>>(Wq, wqb, (int)wsz);
    k_cvt<<<512,  256, 0, stream>>>(Wk, wkb, (int)wsz);
    k_cvt<<<512,  256, 0, stream>>>(Wv, wvb, (int)wsz);
    k_zero<<<(B_BATCH * S_SEQ + 255) / 256, 256, 0, stream>>>(p, B_BATCH * S_SEQ);

    k_qkv<<<B_BATCH * 32, 128, ldsBytes, stream>>>(xb, wqb, wkb, wvb, bq, bk, bv,
                                                   Qb, Kb, Vf);
    k_attn<<<B_BATCH * 32, 128, ldsBytes, stream>>>(Qb, Kb, mask, p);
    k_final<<<B_BATCH, 256, 0, stream>>>(p, Vf, out);
}
